// BiLSTM_CRF_24756191494732
// MI455X (gfx1250) — matrix operand order repaired and checked by reading, not compiled, not run on hardware
//
#include <hip/hip_runtime.h>

typedef _Float16 half_t;
typedef __attribute__((ext_vector_type(16))) _Float16 v16h;
typedef __attribute__((ext_vector_type(8)))  _Float16 v8h;
typedef __attribute__((ext_vector_type(8)))  float    v8f;

#define V_  50000
#define E_  128
#define H_  256
#define L_  7
#define B_  256
#define T_  256
#define G4H 1024   // 4*H

// ---- workspace layout (bytes) ----
#define OFF_X16  ((size_t)0)
#define SZ_X16   ((size_t)2*B_*T_*E_*2)        // 32 MB  (f16 embeds, fwd+rev)
#define OFF_WIH  (OFF_X16 + SZ_X16)
#define SZ_WIH   ((size_t)2*G4H*E_*2)          // 0.5 MB
#define OFF_WHH  (OFF_WIH + SZ_WIH)
#define SZ_WHH   ((size_t)2*G4H*H_*2)          // 1 MB
#define OFF_GX   (OFF_WHH + SZ_WHH)
#define SZ_GX    ((size_t)2*B_*T_*G4H*2)       // 256 MB (f16 x@Wih^T, no bias)
#define OFF_H16  (OFF_GX + SZ_GX)
#define SZ_H16   ((size_t)2*B_*T_*H_*2)        // 64 MB  (f16 h_f, h_b un-reversed)
#define OFF_EM   (OFF_H16 + SZ_H16)            // f32 emissions, padded to 8/token

// ---- gfx1250 async global->LDS path (ASYNCcnt), guarded ----
#if __has_builtin(__builtin_amdgcn_global_load_async_to_lds_b128) && \
    __has_builtin(__builtin_amdgcn_s_wait_asynccnt)
#define HAVE_ASYNC 1
typedef int v4i_t __attribute__((ext_vector_type(4)));
typedef __attribute__((address_space(1))) v4i_t glb_v4i;   // global int32x4
typedef __attribute__((address_space(3))) v4i_t lds_v4i;   // LDS    int32x4
#else
#define HAVE_ASYNC 0
#endif

__device__ __forceinline__ float sigf(float x) { return 1.0f / (1.0f + __expf(-x)); }

// A fragment (16-bit 16x32): lane M = lane&15; V0..3 hold K = hi*8 + 0..7,
// V4..7 hold K = 16 + hi*8 + 0..7  (two contiguous 16B chunks -> b128 loads)
__device__ __forceinline__ v16h load_a_frag(const half_t* base, int rowstride,
                                            int lane, int k0) {
    const half_t* p = base + (size_t)(lane & 15) * rowstride;
    int hi = lane >> 4;
    v8h lo = *(const v8h*)(p + k0 + hi * 8);
    v8h hh = *(const v8h*)(p + k0 + 16 + hi * 8);
    v16h a;
#pragma unroll
    for (int i = 0; i < 8; ++i) { a[i] = lo[i]; a[8 + i] = hh[i]; }
    return a;
}

// B fragment (16-bit 32x16): lane N = lane&15; V0..3 hold K = hi*8 + 0..7,
// V4..7 hold K = 16 + hi*8 + 0..7.  B[k][n] = W[ncol][k]  (W row-major [N,K]).
__device__ __forceinline__ v16h load_b_frag(const half_t* wbase, int rowstride,
                                            int lane, int k0) {
    const half_t* p = wbase + (size_t)(lane & 15) * rowstride;
    int hi = lane >> 4;
    v8h lo = *(const v8h*)(p + k0 + hi * 8);
    v8h hh = *(const v8h*)(p + k0 + 16 + hi * 8);
    v16h b;
#pragma unroll
    for (int i = 0; i < 8; ++i) { b[i] = lo[i]; b[8 + i] = hh[i]; }
    return b;
}

// ---------------- K0: f32 -> f16 weight conversion ----------------
__global__ void __launch_bounds__(256) cvt_f16_kernel(const float* __restrict__ src,
                                                      half_t* __restrict__ dst, int n) {
    int i = blockIdx.x * 256 + threadIdx.x;
    if (i < n) dst[i] = (half_t)src[i];
}

// ---------------- K1: embedding gather (fwd + reversed) ----------------
__global__ void __launch_bounds__(256) gather_kernel(const int* __restrict__ ids,
                                                     const int* __restrict__ lengths,
                                                     const float* __restrict__ table,
                                                     half_t* __restrict__ xf,
                                                     half_t* __restrict__ xb) {
    int i = blockIdx.x * 256 + threadIdx.x;      // over B*T*E
    int e  = i & (E_ - 1);
    int bt = i >> 7;                              // E_ = 128
    int t  = bt & (T_ - 1);
    int b  = bt >> 8;                             // T_ = 256
    xf[i] = (half_t)table[(size_t)ids[bt] * E_ + e];
    int len = lengths[b];
    int tr  = (t < len) ? (len - 1 - t) : t;
    xb[i] = (half_t)table[(size_t)ids[b * T_ + tr] * E_ + e];
}

// ---------------- K2: input GEMM  gx = x @ W_ih^T (both dirs) ----------------
// Grid: (totalTiles/8, 2); 256 threads = 8 waves, 1 wave per 16x16 tile.
// Result tile is transposed through LDS so each lane emits one b128 store.
__global__ void __launch_bounds__(256) input_gemm_kernel(const half_t* __restrict__ ws_x,
                                                         const half_t* __restrict__ ws_wih,
                                                         half_t* __restrict__ ws_gx) {
    __shared__ __align__(16) half_t stage[8][16][16];
    const int tiles_n = G4H / 16;                 // 64
    int dir  = blockIdx.y;
    int wave = threadIdx.x >> 5, lane = threadIdx.x & 31;
    int tile = blockIdx.x * 8 + wave;
    int tm = tile / tiles_n, tn = tile % tiles_n;

    const half_t* x   = ws_x   + (size_t)dir * B_ * T_ * E_;
    const half_t* wih = ws_wih + (size_t)dir * G4H * E_;
    half_t*       gx  = ws_gx  + (size_t)dir * B_ * T_ * G4H;

    const half_t* arow = x + (size_t)tm * 16 * E_;
    const half_t* brow = wih + (size_t)tn * 16 * E_;

    v8f acc = {};
#pragma unroll
    for (int kk = 0; kk < E_ / 32; ++kk) {
        v16h a = load_a_frag(arow, E_, lane, kk * 32);
        v16h b = load_b_frag(brow, E_, lane, kk * 32);
        acc = __builtin_amdgcn_wmma_f32_16x16x32_f16(false, a, false, b,
                                                     (short)0, acc, false, false);
    }
    int hi = lane >> 4, ln = lane & 15;
#pragma unroll
    for (int r = 0; r < 8; ++r) stage[wave][r + 8 * hi][ln] = (half_t)acc[r];
    __syncthreads();
    // each lane stores one contiguous 16B chunk of one output row
    int rr = lane >> 1, cc = lane & 1;
    v8h rowv = *(const v8h*)&stage[wave][rr][cc * 8];
    size_t grow = (size_t)tm * 16 + rr;
    *(v8h*)&gx[grow * G4H + tn * 16 + cc * 8] = rowv;
}

// ---------------- K3: LSTM recurrence (both dirs, 16 batch rows/block) -------
// Grid: 32 blocks (dir*16 + bchunk), 512 threads = 16 waves.
// Wave w owns h-columns [w*16, w*16+16): computes i/f/g/o tiles at
// gate-cols g*256 + w*16, keeps c in registers, h in LDS for next step's A.
// gates_x tiles for step t+1 are async-copied into an LDS double buffer
// while step t's WMMAs run (ASYNCcnt pipeline).
__global__ void __launch_bounds__(512) lstm_rec_kernel(const half_t* __restrict__ ws_gx,
                                                       const half_t* __restrict__ ws_whh,
                                                       half_t* __restrict__ ws_h,
                                                       const float* __restrict__ bihf,
                                                       const float* __restrict__ bhhf,
                                                       const float* __restrict__ bihb,
                                                       const float* __restrict__ bhhb,
                                                       const int* __restrict__ lengths) {
    __shared__ __align__(16) half_t hbuf[16][264];   // 16 rows x 256 (+8 pad)
#if HAVE_ASYNC
    __shared__ __align__(16) half_t gxbuf[2][16][G4H]; // 2 x 32KB double buffer
#endif
    __shared__ int len_s[16];

    int dir   = blockIdx.x >> 4;
    int bbase = (blockIdx.x & 15) * 16;
    int wave  = threadIdx.x >> 5, lane = threadIdx.x & 31;
    int hi = lane >> 4, ln = lane & 15;

    const half_t* gx  = ws_gx  + (size_t)dir * B_ * T_ * G4H;
    const half_t* whh = ws_whh + (size_t)dir * G4H * H_;
    half_t*       hout = ws_h  + (size_t)dir * B_ * T_ * H_;
    const float*  bih = dir ? bihb : bihf;
    const float*  bhh = dir ? bhhb : bhhf;

    for (int i = threadIdx.x; i < 16 * 264; i += 512) (&hbuf[0][0])[i] = (half_t)0.f;
    if (threadIdx.x < 16) len_s[threadIdx.x] = lengths[bbase + threadIdx.x];

    float bias[4];
#pragma unroll
    for (int g = 0; g < 4; ++g) {
        int col = g * H_ + wave * 16 + ln;
        bias[g] = bih[col] + bhh[col];
    }

#if HAVE_ASYNC
    // prologue: async-fill buffer 0 with t=0 tile (16 rows x 2KB, 4 b128/thread)
#pragma unroll
    for (int i = 0; i < 4; ++i) {
        int chunk = threadIdx.x + i * 512;            // 0..2047 (16B chunks)
        int row = chunk >> 7, cc = chunk & 127;
        const half_t* src = gx + ((size_t)(bbase + row) * T_ + 0) * G4H + cc * 8;
        __builtin_amdgcn_global_load_async_to_lds_b128(
            (glb_v4i*)src, (lds_v4i*)&gxbuf[0][row][cc * 8], 0, 0);
    }
    __builtin_amdgcn_s_wait_asynccnt(0);
#endif
    __syncthreads();

    v8f c = {};
    for (int t = 0; t < T_; ++t) {
#if HAVE_ASYNC
        // kick off next step's tile while this step computes
        if (t + 1 < T_) {
            int nb = (t + 1) & 1;
#pragma unroll
            for (int i = 0; i < 4; ++i) {
                int chunk = threadIdx.x + i * 512;
                int row = chunk >> 7, cc = chunk & 127;
                const half_t* src = gx + ((size_t)(bbase + row) * T_ + (t + 1)) * G4H + cc * 8;
                __builtin_amdgcn_global_load_async_to_lds_b128(
                    (glb_v4i*)src, (lds_v4i*)&gxbuf[nb][row][cc * 8], 0, 0);
            }
        }
#endif
        v8f acc[4];
#pragma unroll
        for (int g = 0; g < 4; ++g) {
            int col = g * H_ + wave * 16 + ln;
#pragma unroll
            for (int r = 0; r < 8; ++r) {
#if HAVE_ASYNC
                acc[g][r] = (float)gxbuf[t & 1][r + 8 * hi][col] + bias[g];
#else
                size_t row = (size_t)(bbase + r + 8 * hi) * T_ + t;
                acc[g][r] = (float)gx[row * G4H + col] + bias[g];
#endif
            }
        }
#pragma unroll
        for (int kk = 0; kk < H_ / 32; ++kk) {
            v16h a = load_a_frag(&hbuf[0][0], 264, lane, kk * 32);  // LDS -> ds_load
#pragma unroll
            for (int g = 0; g < 4; ++g) {
                const half_t* brow = whh + (size_t)(g * H_ + wave * 16) * H_;
                v16h b = load_b_frag(brow, H_, lane, kk * 32);      // L2-resident
                acc[g] = __builtin_amdgcn_wmma_f32_16x16x32_f16(false, a, false, b,
                                                                (short)0, acc[g],
                                                                false, false);
            }
        }
        __syncthreads();   // all waves done reading hbuf
#pragma unroll
        for (int r = 0; r < 8; ++r) {
            float ig = sigf(acc[0][r]);
            float fg = sigf(acc[1][r]);
            float gg = tanhf(acc[2][r]);
            float og = sigf(acc[3][r]);
            float cn = fg * c[r] + ig * gg;
            c[r] = cn;
            half_t hv = (half_t)(og * tanhf(cn));
            int row = r + 8 * hi;
            int col = wave * 16 + ln;
            hbuf[row][col] = hv;
            int tpos = t;
            if (dir) { int len = len_s[row]; tpos = (t < len) ? (len - 1 - t) : t; }
            hout[((size_t)(bbase + row) * T_ + tpos) * H_ + col] = hv;
        }
#if HAVE_ASYNC
        __builtin_amdgcn_s_wait_asynccnt(0);  // next buffer fully in LDS
#endif
        __syncthreads();
    }
}

// ---------------- K4: emissions = mask * [h_f,h_b] @ W_lin^T + b_lin ---------
__global__ void __launch_bounds__(256) emis_kernel(const half_t* __restrict__ ws_h,
                                                   const float* __restrict__ wlin,
                                                   const float* __restrict__ blin,
                                                   const int* __restrict__ lengths,
                                                   float* __restrict__ em) {
    __shared__ float wl_s[L_][2 * H_];            // 14 KB
    for (int i = threadIdx.x; i < L_ * 2 * H_; i += 256) (&wl_s[0][0])[i] = wlin[i];
    __syncthreads();

    int bt = blockIdx.x * 256 + threadIdx.x;   // over B*T
    int t = bt & (T_ - 1), b = bt >> 8;
    float acc[L_];
#pragma unroll
    for (int l = 0; l < L_; ++l) acc[l] = blin[l];
    if (t < lengths[b]) {
        const half_t* hf = ws_h + (size_t)bt * H_;
        const half_t* hb = ws_h + (size_t)B_ * T_ * H_ + (size_t)bt * H_;
        for (int j = 0; j < H_; j += 8) {
            v8h vf8 = *(const v8h*)(hf + j);
            v8h vb8 = *(const v8h*)(hb + j);
#pragma unroll
            for (int k = 0; k < 8; ++k) {
                float vf = (float)vf8[k], vb = (float)vb8[k];
#pragma unroll
                for (int l = 0; l < L_; ++l)
                    acc[l] += vf * wl_s[l][j + k] + vb * wl_s[l][H_ + j + k];
            }
        }
    }
#pragma unroll
    for (int l = 0; l < L_; ++l) em[(size_t)bt * 8 + l] = acc[l];
    em[(size_t)bt * 8 + 7] = 0.f;
}

// ---------------- K5: CRF score + log-partition + reduction ------------------
__global__ void __launch_bounds__(256) crf_kernel(const float* __restrict__ em,
                                                  const int* __restrict__ labels,
                                                  const int* __restrict__ lengths,
                                                  const float* __restrict__ start_t,
                                                  const float* __restrict__ end_t,
                                                  const float* __restrict__ trans,
                                                  float* __restrict__ out) {
    __shared__ float tr_s[L_][L_];
    __shared__ float st_s[L_], en_s[L_];
    __shared__ float red[256];
    int tid = threadIdx.x;
    if (tid < L_ * L_) tr_s[tid / L_][tid % L_] = trans[tid];
    if (tid < L_) { st_s[tid] = start_t[tid]; en_s[tid] = end_t[tid]; }
    __syncthreads();

    int b = tid;
    int len = lengths[b];
    const int*   tg = labels + b * T_;
    const float* eb = em + (size_t)b * T_ * 8;

    float alpha[L_];
    int prev = tg[0];
    float score = st_s[prev] + eb[prev];
#pragma unroll
    for (int j = 0; j < L_; ++j) alpha[j] = st_s[j] + eb[j];

    for (int t = 1; t < T_; ++t) {
        int cur = tg[t];
        if (t < len) {
            score += tr_s[prev][cur] + eb[t * 8 + cur];
            float nxt[L_];
#pragma unroll
            for (int j = 0; j < L_; ++j) {
                float m = -1e30f;
#pragma unroll
                for (int i = 0; i < L_; ++i) m = fmaxf(m, alpha[i] + tr_s[i][j]);
                float s = 0.f;
#pragma unroll
                for (int i = 0; i < L_; ++i) s += __expf(alpha[i] + tr_s[i][j] - m);
                nxt[j] = m + __logf(s) + eb[t * 8 + j];
            }
#pragma unroll
            for (int j = 0; j < L_; ++j) alpha[j] = nxt[j];
        }
        prev = cur;
    }
    score += en_s[tg[len - 1]];

    float m = -1e30f;
#pragma unroll
    for (int j = 0; j < L_; ++j) m = fmaxf(m, alpha[j] + en_s[j]);
    float s = 0.f;
#pragma unroll
    for (int j = 0; j < L_; ++j) s += __expf(alpha[j] + en_s[j] - m);
    float logz = m + __logf(s);

    red[tid] = score - logz;
    __syncthreads();
    for (int off = 128; off > 0; off >>= 1) {
        if (tid < off) red[tid] += red[tid + off];
        __syncthreads();
    }
    if (tid == 0) out[0] = -red[0];
}

extern "C" void kernel_launch(void* const* d_in, const int* in_sizes, int n_in,
                              void* d_out, int out_size, void* d_ws, size_t ws_size,
                              hipStream_t stream) {
    const int*   input_ids = (const int*)  d_in[0];
    const int*   lengths   = (const int*)  d_in[1];
    const int*   label_ids = (const int*)  d_in[2];
    const float* embed     = (const float*)d_in[3];
    const float* w_ih_f    = (const float*)d_in[4];
    const float* w_hh_f    = (const float*)d_in[5];
    const float* b_ih_f    = (const float*)d_in[6];
    const float* b_hh_f    = (const float*)d_in[7];
    const float* w_ih_b    = (const float*)d_in[8];
    const float* w_hh_b    = (const float*)d_in[9];
    const float* b_ih_b    = (const float*)d_in[10];
    const float* b_hh_b    = (const float*)d_in[11];
    const float* w_lin     = (const float*)d_in[12];
    const float* b_lin     = (const float*)d_in[13];
    const float* start_tr  = (const float*)d_in[14];
    const float* end_tr    = (const float*)d_in[15];
    const float* trans     = (const float*)d_in[16];

    char* ws = (char*)d_ws;
    half_t* x16  = (half_t*)(ws + OFF_X16);
    half_t* wih  = (half_t*)(ws + OFF_WIH);
    half_t* whh  = (half_t*)(ws + OFF_WHH);
    half_t* gx   = (half_t*)(ws + OFF_GX);
    half_t* h16  = (half_t*)(ws + OFF_H16);
    float*  emis = (float*)(ws + OFF_EM);

    // K0: weight conversions (f32 -> f16)
    cvt_f16_kernel<<<(G4H * E_ + 255) / 256, 256, 0, stream>>>(w_ih_f, wih, G4H * E_);
    cvt_f16_kernel<<<(G4H * E_ + 255) / 256, 256, 0, stream>>>(w_ih_b, wih + G4H * E_, G4H * E_);
    cvt_f16_kernel<<<(G4H * H_ + 255) / 256, 256, 0, stream>>>(w_hh_f, whh, G4H * H_);
    cvt_f16_kernel<<<(G4H * H_ + 255) / 256, 256, 0, stream>>>(w_hh_b, whh + G4H * H_, G4H * H_);

    // K1: embedding gather (fwd + reversed)
    gather_kernel<<<(B_ * T_ * E_) / 256, 256, 0, stream>>>(
        input_ids, lengths, embed, x16, x16 + (size_t)B_ * T_ * E_);

    // K2: input GEMM for both directions
    {
        dim3 grid((B_ * T_ / 16) * (G4H / 16) / 8, 2);
        input_gemm_kernel<<<grid, 256, 0, stream>>>(x16, wih, gx);
    }

    // K3: LSTM recurrence (both directions)
    lstm_rec_kernel<<<32, 512, 0, stream>>>(gx, whh, h16,
                                            b_ih_f, b_hh_f, b_ih_b, b_hh_b, lengths);

    // K4: emissions
    emis_kernel<<<(B_ * T_) / 256, 256, 0, stream>>>(h16, w_lin, b_lin, lengths, emis);

    // K5: CRF loss
    crf_kernel<<<1, 256, 0, stream>>>(emis, label_ids, lengths,
                                      start_tr, end_tr, trans, (float*)d_out);
}
